// EdgeAttention_77249281785953
// MI455X (gfx1250) — matrix operand order repaired and checked by reading, not compiled, not run on hardware
//
#include <hip/hip_runtime.h>

typedef __attribute__((ext_vector_type(16))) _Float16 v16h;
typedef __attribute__((ext_vector_type(8)))  _Float16 v8h;
typedef __attribute__((ext_vector_type(8)))  float    v8f;

#define C_DIM 256
#define BATCH 8
#define BM 128
#define BN 128
#define BK 32
#define KSTEPS (C_DIM / BK)
#define KPAD 40   // LDS row stride in halves: 80B keeps 16B alignment for b128 ops

// ---------------------------------------------------------------------------
// CDNA5 async global->LDS copy (ASYNCcnt-tracked), 16B per lane.
// LDS address comes from the VDST VGPR; low 32 bits of a flat shared pointer are the
// LDS byte offset.
// ---------------------------------------------------------------------------
__device__ __forceinline__ void async_b128(const void* gsrc, void* lds_dst) {
    unsigned int       l = (unsigned int)(unsigned long long)lds_dst;
    unsigned long long g = (unsigned long long)gsrc;
    asm volatile("global_load_async_to_lds_b128 %0, %1, off"
                 :: "v"(l), "v"(g) : "memory");
}
__device__ __forceinline__ void wait_async_le2() {
    asm volatile("s_wait_asynccnt 0x2" ::: "memory");
}
__device__ __forceinline__ void wait_async_0() {
    asm volatile("s_wait_asynccnt 0x0" ::: "memory");
}

// ---------------------------------------------------------------------------
// Prepack 1: W [C,C] f32 -> f16 (same layout). 65536 elems per scale.
// ---------------------------------------------------------------------------
__global__ void pack_w_f16(const float* __restrict__ W, _Float16* __restrict__ Wh) {
    const int i = blockIdx.x * 256 + threadIdx.x;
    Wh[i] = (_Float16)W[i];
}

// ---------------------------------------------------------------------------
// Prepack 2: F [B,C,L] f32 -> Ft [B,L,C] f16 (tiled 32x32 LDS transpose,
// coalesced on both sides). Makes K (=channels) contiguous for async staging.
// ---------------------------------------------------------------------------
__global__ __launch_bounds__(256)
void transpose_f16(const float* __restrict__ F, _Float16* __restrict__ Ft, int L) {
    __shared__ float t[32][33];
    const int b  = blockIdx.z;
    const int l0 = blockIdx.x * 32;
    const int c0 = blockIdx.y * 32;
    const int lx = threadIdx.x & 31;
    const int cy = threadIdx.x >> 5;   // 0..7
#pragma unroll
    for (int it = 0; it < 4; ++it) {
        const int c = c0 + cy + it * 8;
        const int l = l0 + lx;
        t[cy + it * 8][lx] = (l < L) ? F[((size_t)b * C_DIM + c) * L + l] : 0.0f;
    }
    __syncthreads();
#pragma unroll
    for (int it = 0; it < 4; ++it) {
        const int l = l0 + cy + it * 8;
        const int c = c0 + lx;
        if (l < L) Ft[((size_t)b * L + l) * C_DIM + c] = (_Float16)t[lx][cy + it * 8];
    }
}

// ---------------------------------------------------------------------------
// Phase 1: P[b,o,l] = sum_c Wh[o,c]*Ft[b,l,c] + bias[o]  (f16 in, f16 out).
// 256 threads = 8 waves (wave32). Block tile 128x128x32, wave tile 64x32.
// Double-buffered LDS staged with GLOBAL_LOAD_ASYNC_TO_LDS_B128:
// each thread issues exactly 2 async b128s per K-step; next stage is in
// flight while the current stage's 8 WMMAs execute.
// ---------------------------------------------------------------------------
__global__ __launch_bounds__(256)
void proj_gemm_async(const _Float16* __restrict__ Wh,   // [C,C]   f16
                     const _Float16* __restrict__ Ft,   // [B,L,C] f16
                     const float*    __restrict__ bias, // [C]     f32
                     _Float16*       __restrict__ P,    // [B,C,L] f16 out
                     int L)
{
    __shared__ _Float16 As[2][BM * KPAD];   // W tile  [m][k]
    __shared__ _Float16 Bs[2][BN * KPAD];   // Ft tile [n][k]
    __shared__ float    BiasS[BM];

    const int tid  = threadIdx.x;
    const int lane = tid & 31;
    const int wave = tid >> 5;
    const int wm   = wave & 1;      // wave M tile: wm*64
    const int wn   = wave >> 1;     // wave N tile: wn*32
    const int llo  = lane & 15;
    const int kg   = lane >> 4;

    const int m0 = blockIdx.y * BM;
    const int n0 = blockIdx.x * BN;
    const int b  = blockIdx.z;

    if (tid < BM) BiasS[tid] = bias[m0 + tid];

    // per-thread async-copy assignment: 16B chunk (row, half-of-row)
    const int crow = tid >> 1;           // 0..127
    const int ckq  = (tid & 1) * 16;     // 0 or 16 halves
    const _Float16* aSrc = Wh + (size_t)(m0 + crow) * C_DIM + ckq;
    // Rows n>=L read harmless garbage from later workspace regions; a B column
    // only feeds its own (never stored) output column, so no guard is needed.
    const _Float16* bSrc = Ft + ((size_t)b * L + (size_t)(n0 + crow)) * C_DIM + ckq;
    const unsigned  ldsOff = crow * KPAD + ckq;

    v8f acc[4][2];
#pragma unroll
    for (int i = 0; i < 4; ++i)
#pragma unroll
        for (int j = 0; j < 2; ++j) acc[i][j] = v8f{};

    // prologue: stage 0 into buffer 0
    async_b128(aSrc, &As[0][ldsOff]);
    async_b128(bSrc, &Bs[0][ldsOff]);

    for (int k = 0; k < KSTEPS; ++k) {
        const int cur = k & 1;
        if (k + 1 < KSTEPS) {
            const int nxt = (k + 1) & 1;
            async_b128(aSrc + (k + 1) * BK, &As[nxt][ldsOff]);
            async_b128(bSrc + (k + 1) * BK, &Bs[nxt][ldsOff]);
            wait_async_le2();            // stage k landed (in-order completion)
        } else {
            wait_async_0();
        }
        __syncthreads();                 // all waves' stage-k copies visible

        // A 16x32 frag: row=llo; halves 0..7 -> K=kg*8+0..7, 8..15 -> K=16+kg*8+0..7
        v16h afr[4];
#pragma unroll
        for (int i = 0; i < 4; ++i) {
            const _Float16* src = &As[cur][(wm * 64 + i * 16 + llo) * KPAD + kg * 8];
            v8h lo = *(const v8h*)(src);
            v8h hi = *(const v8h*)(src + 16);
            afr[i] = __builtin_shufflevector(lo, hi, 0,1,2,3,4,5,6,7,8,9,10,11,12,13,14,15);
        }
        // B 32x16 frag: col=llo; halves 0..7 -> K=kg*8+0..7, 8..15 -> K=16+kg*8+0..7
        v16h bfr[2];
#pragma unroll
        for (int j = 0; j < 2; ++j) {
            const _Float16* src = &Bs[cur][(wn * 32 + j * 16 + llo) * KPAD + kg * 8];
            v8h lo = *(const v8h*)(src);
            v8h hi = *(const v8h*)(src + 16);
            bfr[j] = __builtin_shufflevector(lo, hi, 0,1,2,3,4,5,6,7,8,9,10,11,12,13,14,15);
        }
#pragma unroll
        for (int i = 0; i < 4; ++i)
#pragma unroll
            for (int j = 0; j < 2; ++j)
                acc[i][j] = __builtin_amdgcn_wmma_f32_16x16x32_f16(
                    false, afr[i], false, bfr[j], (short)0, acc[i][j], false, false);

        __syncthreads();                 // protect buffers before next async issue
    }

    // epilogue: bias add, f32->f16, store (C frag: VGPR r -> M=kg*8+r, N=llo)
    _Float16* Pb = P + (size_t)b * C_DIM * L;
#pragma unroll
    for (int i = 0; i < 4; ++i) {
#pragma unroll
        for (int j = 0; j < 2; ++j) {
            const int ncol = n0 + wn * 32 + j * 16 + llo;
            if (ncol < L) {
#pragma unroll
                for (int r = 0; r < 8; ++r) {
                    const int o = m0 + wm * 64 + i * 16 + kg * 8 + r;
                    Pb[(size_t)o * L + ncol] = (_Float16)(acc[i][j][r] + BiasS[o - m0]);
                }
            }
        }
    }
}

// ---------------------------------------------------------------------------
__global__ void zero_acc(float* __restrict__ g) { g[threadIdx.x] = 0.0f; }

// ---------------------------------------------------------------------------
// Phase 2: per-pixel cosine similarity across the 4 scales with nearest-
// neighbor gather (upsampling never materialized). One thread per pixel,
// wave32 shuffle reduce + f32 atomics into gacc[b*16 + i*4 + j].
// ---------------------------------------------------------------------------
__global__ __launch_bounds__(256)
void cos_accum(const _Float16* __restrict__ p0, const _Float16* __restrict__ p1,
               const _Float16* __restrict__ p2, const _Float16* __restrict__ p3,
               float* __restrict__ gacc)
{
    const int b = blockIdx.z;
    const int l = blockIdx.x * 256 + threadIdx.x;   // 25 blocks * 256 = 6400
    const int y = l / 80, x = l % 80;

    const int L0 = 6400, L1 = 1600, L2 = 400, L3 = 100;
    const _Float16* v0 = p0 + (size_t)b * C_DIM * L0 + (y * 80 + x);
    const _Float16* v1 = p1 + (size_t)b * C_DIM * L1 + ((y >> 1) * 40 + (x >> 1));
    const _Float16* v2 = p2 + (size_t)b * C_DIM * L2 + ((y >> 2) * 20 + (x >> 2));
    const _Float16* v3 = p3 + (size_t)b * C_DIM * L3 + ((y >> 3) * 10 + (x >> 3));

    float d00 = 0.f, d01 = 0.f, d02 = 0.f, d03 = 0.f;
    float d11 = 0.f, d12 = 0.f, d13 = 0.f;
    float d22 = 0.f, d23 = 0.f, d33 = 0.f;

#pragma unroll 4
    for (int c = 0; c < C_DIM; ++c) {
        const float a0 = (float)v0[(size_t)c * L0];
        const float a1 = (float)v1[(size_t)c * L1];
        const float a2 = (float)v2[(size_t)c * L2];
        const float a3 = (float)v3[(size_t)c * L3];
        d00 += a0 * a0; d01 += a0 * a1; d02 += a0 * a2; d03 += a0 * a3;
        d11 += a1 * a1; d12 += a1 * a2; d13 += a1 * a3;
        d22 += a2 * a2; d23 += a2 * a3;
        d33 += a3 * a3;
    }

    float dd[4][4];
    dd[0][0] = d00; dd[0][1] = d01; dd[0][2] = d02; dd[0][3] = d03;
    dd[1][0] = d01; dd[1][1] = d11; dd[1][2] = d12; dd[1][3] = d13;
    dd[2][0] = d02; dd[2][1] = d12; dd[2][2] = d22; dd[2][3] = d23;
    dd[3][0] = d03; dd[3][1] = d13; dd[3][2] = d23; dd[3][3] = d33;

    float nn[4];
#pragma unroll
    for (int i = 0; i < 4; ++i) nn[i] = sqrtf(dd[i][i]);

    float cs[16];
#pragma unroll
    for (int i = 0; i < 4; ++i)
#pragma unroll
        for (int j = 0; j < 4; ++j)
            cs[i * 4 + j] = dd[i][j] / fmaxf(nn[i] * nn[j], 1e-8f);

#pragma unroll
    for (int k = 0; k < 16; ++k) {
        float v = cs[k];
        for (int off = 16; off > 0; off >>= 1) v += __shfl_down(v, off, 32);
        cs[k] = v;
    }
    if ((threadIdx.x & 31) == 0) {
#pragma unroll
        for (int k = 0; k < 16; ++k) atomicAdd(&gacc[b * 16 + k], cs[k]);
    }
}

// ---------------------------------------------------------------------------
// Phase 3: mean + row softmax. 128 threads = [B=8][i=4][j=4]; rows are
// groups of 4 consecutive lanes -> xor shuffles.
// ---------------------------------------------------------------------------
__global__ void softmax_out(const float* __restrict__ gacc, float* __restrict__ out)
{
    const int t = threadIdx.x;           // 0..127
    const float a = gacc[t] * (1.0f / 6400.0f);
    float m = a;
    m = fmaxf(m, __shfl_xor(m, 1, 32));
    m = fmaxf(m, __shfl_xor(m, 2, 32));
    const float e = __expf(a - m);
    float s = e;
    s += __shfl_xor(s, 1, 32);
    s += __shfl_xor(s, 2, 32);
    out[t] = e / s;
}

// ---------------------------------------------------------------------------
extern "C" void kernel_launch(void* const* d_in, const int* in_sizes, int n_in,
                              void* d_out, int out_size, void* d_ws, size_t ws_size,
                              hipStream_t stream)
{
    (void)in_sizes; (void)n_in; (void)out_size; (void)ws_size;
    const float* F[4]; const float* W[4]; const float* Bv[4];
    for (int i = 0; i < 4; ++i) {
        F[i]  = (const float*)d_in[3 * i + 0];
        W[i]  = (const float*)d_in[3 * i + 1];
        Bv[i] = (const float*)d_in[3 * i + 2];
    }
    const int Ls[4] = {6400, 1600, 400, 100};

    char* ws = (char*)d_ws;
    size_t off = 0;
    auto take = [&](size_t bytes) { char* p = ws + off; off = (off + bytes + 255) & ~(size_t)255; return p; };

    _Float16* Wh[4]; _Float16* Ft[4]; _Float16* P[4];
    for (int i = 0; i < 4; ++i) Wh[i] = (_Float16*)take((size_t)C_DIM * C_DIM * sizeof(_Float16));
    for (int i = 0; i < 4; ++i) Ft[i] = (_Float16*)take((size_t)BATCH * C_DIM * Ls[i] * sizeof(_Float16));
    for (int i = 0; i < 4; ++i) P[i]  = (_Float16*)take((size_t)BATCH * C_DIM * Ls[i] * sizeof(_Float16));
    float* gacc = (float*)take(128 * sizeof(float));

    for (int i = 0; i < 4; ++i)
        pack_w_f16<<<dim3((C_DIM * C_DIM) / 256), dim3(256), 0, stream>>>(W[i], Wh[i]);
    for (int i = 0; i < 4; ++i)
        transpose_f16<<<dim3((Ls[i] + 31) / 32, C_DIM / 32, BATCH), dim3(256), 0, stream>>>(F[i], Ft[i], Ls[i]);
    for (int i = 0; i < 4; ++i)
        proj_gemm_async<<<dim3((Ls[i] + BN - 1) / BN, C_DIM / BM, BATCH), dim3(256), 0, stream>>>(
            Wh[i], Ft[i], Bv[i], P[i], Ls[i]);
    zero_acc<<<dim3(1), dim3(128), 0, stream>>>(gacc);
    cos_accum<<<dim3(25, 1, BATCH), dim3(256), 0, stream>>>(P[0], P[1], P[2], P[3], gacc);
    softmax_out<<<dim3(1), dim3(128), 0, stream>>>(gacc, (float*)d_out);
}
